// InverseLensLayer_9586367004692
// MI455X (gfx1250) — matrix operand order repaired and checked by reading, not compiled, not run on hardware
//
#include <hip/hip_runtime.h>
#include <hip/hip_bf16.h>
#include <math.h>

typedef __attribute__((ext_vector_type(16))) _Float16 v16h;
typedef __attribute__((ext_vector_type(8)))  float    v8f;

#define BB 64
#define HH 128
#define WW 128
#define HW 16384  // 128*128

// ---------------------------------------------------------------------------
// 1) Build x16 = concat(image, r, cos(phi), sin(phi), 12 x zero)  [B,16,H,W]
//    (channel padding to CP=16 so layer-1 convs use the uniform WMMA path)
// ---------------------------------------------------------------------------
__global__ void __launch_bounds__(256) build_x16_kernel(const float* __restrict__ img,
                                                        float* __restrict__ X) {
    int i = blockIdx.x * 256 + threadIdx.x;          // over B*HW
    if (i >= BB * HW) return;
    int b = i >> 14;
    int p = i & (HW - 1);
    int py = p >> 7, px = p & 127;
    float xc = -1.0f + 2.0f * (float)px / 127.0f;
    float yc = -1.0f + 2.0f * (float)py / 127.0f;
    float r  = sqrtf(xc * xc + yc * yc);
    float ph = atan2f(yc, xc);
    size_t base = (size_t)b * 16 * HW + p;
    X[base + 0 * HW] = img[i];
    X[base + 1 * HW] = r;
    X[base + 2 * HW] = cosf(ph);
    X[base + 3 * HW] = sinf(ph);
    #pragma unroll
    for (int c = 4; c < 16; ++c) X[base + (size_t)c * HW] = 0.0f;
}

// ---------------------------------------------------------------------------
// 2) Weight pre-pack: OIHW f32 -> per-lane A-fragment f16 stream.
//    GEMM K ordering: k = tap*CP + cin  (tap = ky*3+kx, CP = padded channels).
//    A fragment (contiguous 16x32 layout): element e -> k = 32c+(hi?16:0)+e.
//    Zero for tap>=9 or cin>=CINreal.
// ---------------------------------------------------------------------------
__global__ void __launch_bounds__(256) pack_w_kernel(const float* __restrict__ w,
                                                     _Float16* __restrict__ wp,
                                                     int CINreal, int CP, int L2CP,
                                                     int MT, int NCH) {
    int i = blockIdx.x * 256 + threadIdx.x;
    int tot = MT * NCH * 32 * 16;
    if (i >= tot) return;
    int e    = i & 15;
    int lane = (i >> 4) & 31;
    int cm   = i >> 9;               // mt*NCH + c
    int c    = cm % NCH;
    int mt   = cm / NCH;
    int m    = mt * 16 + (lane & 15);
    int k    = 32 * c + ((lane >= 16) ? 16 : 0) + e;
    int r    = k >> L2CP;
    int cin  = k & (CP - 1);
    float v = 0.0f;
    if (r < 9 && cin < CINreal) v = w[(m * CINreal + cin) * 9 + r];
    wp[i] = (_Float16)v;
}

// ---------------------------------------------------------------------------
// 3) 3x3 conv (pad=1), LDS-staged implicit GEMM on v_wmma_f32_16x16x32_f16.
//    Block: 8 waves; output tile 8 rows x 16 cols; wave w -> row y0+w.
//    LDS layout: lds[row][col][cin] (channel-minor) so each K-half of the
//    B fragment is one contiguous, 32B-aligned ds_read.
// ---------------------------------------------------------------------------
template<int CP, int L2CP, int COUT, int NCH, int NROWS>
__global__ void __launch_bounds__(256) conv3x3_lds_wmma(
    const float*   __restrict__ in,     // [B,CP,H,W]
    const _Float16* __restrict__ wpack, // packed A fragments
    const float*   __restrict__ bias,   // [COUT]
    float*         __restrict__ out)    // [B,COUT,H,W]
{
    __shared__ __align__(16) _Float16 lds[NROWS * 18 * CP];
    const int tid = threadIdx.x;
    const int xt = blockIdx.x & 7, yt = blockIdx.x >> 3;   // 8 col-tiles, 16 row-tiles
    const int x0 = xt * 16, y0 = yt * 8;
    const int mTile = blockIdx.y;
    const int b     = blockIdx.z;

    // ---- stage input tile (halo, zero borders), coalesced along x ----
    const float* inb = in + (size_t)b * CP * HW;
    const int TOT = CP * NROWS * 18;
    for (int i = tid; i < TOT; i += 256) {
        int col = i % 18;
        int t   = i / 18;
        int row = t % NROWS;
        int cin = t / NROWS;
        int gx = x0 - 1 + col, gy = y0 - 1 + row;
        float v = 0.0f;
        if (gx >= 0 && gx < WW && gy >= 0 && gy < HH)
            v = inb[(size_t)cin * HW + (gy << 7) + gx];
        lds[(row * 18 + col) * CP + cin] = (_Float16)v;
    }
    __syncthreads();

    const int wave = tid >> 5;
    const int lane = tid & 31;
    const bool hi  = lane >= 16;
    const int n    = lane & 15;

    const v16h* wp = (const v16h*)wpack + (size_t)(mTile * NCH) * 32 + lane;

    v8f acc = {};
    #pragma unroll
    for (int c = 0; c < NCH; ++c) {
        v16h A = wp[c * 32];                       // 2x global_load_b128
        int addr;
        if (CP == 32) {
            // chunk c == tap c; lo lanes cin 0..15, hi lanes cin 16..31
            const int ky = c / 3 - 1, kx = c % 3 - 1;
            addr = ((wave + ky + 1) * 18 + (n + kx + 1)) * CP + (hi ? 16 : 0);
        } else {
            // CP==16: lo lanes tap 2c, hi lanes tap 2c+1; full 16-ch run each
            const int r0 = 2 * c, r1 = 2 * c + 1;
            const int a0 = ((wave + r0 / 3) * 18 + (n + (r0 % 3))) * CP;
            const int a1 = ((wave + r1 / 3) * 18 + (n + (r1 % 3))) * CP;
            addr = hi ? a1 : a0;
        }
        v16h Bm = *(const v16h*)(lds + addr);      // 2x ds_read_b128
        acc = __builtin_amdgcn_wmma_f32_16x16x32_f16(
            false, A, false, Bm, (short)0, acc, false, false);
    }

    const int pix = ((y0 + wave) << 7) + x0 + n;
    #pragma unroll
    for (int r = 0; r < 8; ++r) {
        int m = mTile * 16 + r + (hi ? 8 : 0);
        out[((size_t)b * COUT + m) * HW + pix] = acc[r] + bias[m];
    }
}

// ---------------------------------------------------------------------------
// 4) GroupNorm statistics: one block per (batch, group)
// ---------------------------------------------------------------------------
__global__ void __launch_bounds__(256) gn_stats_kernel(const float* __restrict__ buf,
                                                       float* __restrict__ stats,
                                                       int C, int groups) {
    int bg = blockIdx.x;
    int b  = bg / groups, g = bg % groups;
    int Cg = C / groups;
    int n  = Cg * HW;
    const float* base = buf + ((size_t)b * C + (size_t)g * Cg) * HW;
    float s = 0.f, s2 = 0.f;
    for (int i = threadIdx.x; i < n; i += 256) {
        float v = base[i];
        s += v; s2 += v * v;
    }
    __shared__ float sh[512];
    sh[threadIdx.x] = s; sh[256 + threadIdx.x] = s2;
    __syncthreads();
    for (int o = 128; o > 0; o >>= 1) {
        if (threadIdx.x < o) {
            sh[threadIdx.x]       += sh[threadIdx.x + o];
            sh[256 + threadIdx.x] += sh[256 + threadIdx.x + o];
        }
        __syncthreads();
    }
    if (threadIdx.x == 0) {
        float mu  = sh[0] / (float)n;
        float var = sh[256] / (float)n - mu * mu;
        stats[2 * bg]     = mu;
        stats[2 * bg + 1] = rsqrtf(var + 1e-5f);
    }
}

// ---------------------------------------------------------------------------
// 5) Apply GroupNorm + SiLU
// ---------------------------------------------------------------------------
__global__ void __launch_bounds__(256) gn_silu_kernel(const float* __restrict__ in,
                                                      float* __restrict__ out,
                                                      const float* __restrict__ gamma,
                                                      const float* __restrict__ beta,
                                                      const float* __restrict__ stats,
                                                      int C, int groups) {
    size_t i = (size_t)blockIdx.x * 256 + threadIdx.x;
    size_t total = (size_t)BB * C * HW;
    if (i >= total) return;
    int c = (int)((i >> 14) % (size_t)C);
    int b = (int)(i / ((size_t)C << 14));
    int g = c / (C / groups);
    float mu = stats[2 * (b * groups + g)];
    float rs = stats[2 * (b * groups + g) + 1];
    float xn = (in[i] - mu) * rs * gamma[c] + beta[c];
    out[i] = xn / (1.0f + expf(-xn));  // SiLU
}

// ---------------------------------------------------------------------------
// 6) k head: 1x1 conv (16->1) + scaled tanh
// ---------------------------------------------------------------------------
__global__ void __launch_bounds__(256) head_k_kernel(const float* __restrict__ a,
                                                     const float* __restrict__ w,
                                                     const float* __restrict__ bb,
                                                     float* __restrict__ kout) {
    int i = blockIdx.x * 256 + threadIdx.x;
    if (i >= BB * HW) return;
    int b = i >> 14, p = i & (HW - 1);
    float s = bb[0];
    #pragma unroll
    for (int c = 0; c < 16; ++c)
        s += w[c] * a[((size_t)b * 16 + c) * HW + p];
    kout[i] = 0.5f * (1.0f + 0.3f * tanhf(s));
}

// ---------------------------------------------------------------------------
// 7) psi head: psi = k*r + 0.05*tanh(1x1conv(a))
// ---------------------------------------------------------------------------
__global__ void __launch_bounds__(256) head_psi_kernel(const float* __restrict__ a,
                                                       const float* __restrict__ w,
                                                       const float* __restrict__ bb,
                                                       const float* __restrict__ kin,
                                                       float* __restrict__ psi) {
    int i = blockIdx.x * 256 + threadIdx.x;
    if (i >= BB * HW) return;
    int b = i >> 14, p = i & (HW - 1);
    int py = p >> 7, px = p & 127;
    float xc = -1.0f + 2.0f * (float)px / 127.0f;
    float yc = -1.0f + 2.0f * (float)py / 127.0f;
    float r  = sqrtf(xc * xc + yc * yc);
    float s = bb[0];
    #pragma unroll
    for (int c = 0; c < 16; ++c)
        s += w[c] * a[((size_t)b * 16 + c) * HW + p];
    psi[i] = kin[i] * r + 0.05f * tanhf(s);
}

// ---------------------------------------------------------------------------
// 8) Separable 5-tap Gaussian blur with reflect padding
// ---------------------------------------------------------------------------
__device__ __forceinline__ void gauss_w(float* kw) {
    float s = 0.f;
    #pragma unroll
    for (int j = 0; j < 5; ++j) { float o = (float)j - 2.0f; kw[j] = expf(-o * o * 0.5f); s += kw[j]; }
    #pragma unroll
    for (int j = 0; j < 5; ++j) kw[j] /= s;
}

__global__ void __launch_bounds__(256) blur_h_kernel(const float* __restrict__ in,
                                                     float* __restrict__ out) {
    int i = blockIdx.x * 256 + threadIdx.x;
    if (i >= BB * HW) return;
    int b = i >> 14, p = i & (HW - 1);
    int py = p >> 7, px = p & 127;
    float kw[5]; gauss_w(kw);
    const float* row = in + (size_t)b * HW + (py << 7);
    float s = 0.f;
    #pragma unroll
    for (int t = -2; t <= 2; ++t) {
        int xi = px + t;
        if (xi < 0) xi = -xi;
        if (xi > 127) xi = 254 - xi;
        s += kw[t + 2] * row[xi];
    }
    out[i] = s;
}

__global__ void __launch_bounds__(256) blur_v_kernel(const float* __restrict__ in,
                                                     float* __restrict__ out) {
    int i = blockIdx.x * 256 + threadIdx.x;
    if (i >= BB * HW) return;
    int b = i >> 14, p = i & (HW - 1);
    int py = p >> 7, px = p & 127;
    float kw[5]; gauss_w(kw);
    const float* img = in + (size_t)b * HW;
    float s = 0.f;
    #pragma unroll
    for (int t = -2; t <= 2; ++t) {
        int yi = py + t;
        if (yi < 0) yi = -yi;
        if (yi > 127) yi = 254 - yi;
        s += kw[t + 2] * img[(yi << 7) + px];
    }
    out[i] = s;
}

// ---------------------------------------------------------------------------
// 9) Gradients, soft clamp, grid-sample warp (border), blend
// ---------------------------------------------------------------------------
__global__ void __launch_bounds__(256) final_kernel(const float* __restrict__ psis,
                                                    const float* __restrict__ img,
                                                    float* __restrict__ src,
                                                    float* __restrict__ ax_o,
                                                    float* __restrict__ ay_o) {
    int i = blockIdx.x * 256 + threadIdx.x;
    if (i >= BB * HW) return;
    int b = i >> 14, p = i & (HW - 1);
    int py = p >> 7, px = p & 127;
    const float* ps = psis + (size_t)b * HW;
    const float dx = 2.0f / 127.0f;

    float gx, gy;
    if (px == 0)        gx = ps[(py << 7) + 1] - ps[(py << 7)];
    else if (px == 127) gx = ps[(py << 7) + 127] - ps[(py << 7) + 126];
    else                gx = 0.5f * (ps[(py << 7) + px + 1] - ps[(py << 7) + px - 1]);
    if (py == 0)        gy = ps[(1 << 7) + px] - ps[px];
    else if (py == 127) gy = ps[(127 << 7) + px] - ps[(126 << 7) + px];
    else                gy = 0.5f * (ps[((py + 1) << 7) + px] - ps[((py - 1) << 7) + px]);
    gx /= dx; gy /= dx;

    float ax = 0.5f * tanhf(gx / 0.5f);
    float ay = 0.5f * tanhf(gy / 0.5f);

    float X = -1.0f + 2.0f * (float)px / 127.0f;
    float Y = -1.0f + 2.0f * (float)py / 127.0f;
    float bx = fminf(fmaxf(X - ax, -1.0f), 1.0f);
    float by = fminf(fmaxf(Y - ay, -1.0f), 1.0f);

    float pxf = (bx + 1.0f) * 0.5f * 127.0f;
    float pyf = (by + 1.0f) * 0.5f * 127.0f;
    float x0f = floorf(pxf), y0f = floorf(pyf);
    float wx = pxf - x0f, wy = pyf - y0f;
    int x0 = min(max((int)x0f, 0), 127);
    int x1 = min(x0 + 1, 127);
    int y0 = min(max((int)y0f, 0), 127);
    int y1 = min(y0 + 1, 127);
    const float* im = img + (size_t)b * HW;
    float v00 = im[(y0 << 7) + x0], v01 = im[(y0 << 7) + x1];
    float v10 = im[(y1 << 7) + x0], v11 = im[(y1 << 7) + x1];
    float warped = v00 * (1 - wx) * (1 - wy) + v01 * wx * (1 - wy)
                 + v10 * (1 - wx) * wy       + v11 * wx * wy;

    src[i]  = 0.9f * warped + 0.1f * im[p];
    ax_o[i] = ax;
    ay_o[i] = ay;
}

// ---------------------------------------------------------------------------
// Orchestration
// ---------------------------------------------------------------------------
extern "C" void kernel_launch(void* const* d_in, const int* in_sizes, int n_in,
                              void* d_out, int out_size, void* d_ws, size_t ws_size,
                              hipStream_t stream) {
    const float* image  = (const float*)d_in[0];
    const float* kw1 = (const float*)d_in[1];  const float* kb1 = (const float*)d_in[2];
    const float* kg1 = (const float*)d_in[3];  const float* kbe1= (const float*)d_in[4];
    const float* kw2 = (const float*)d_in[5];  const float* kb2 = (const float*)d_in[6];
    const float* kg2 = (const float*)d_in[7];  const float* kbe2= (const float*)d_in[8];
    const float* kw3 = (const float*)d_in[9];  const float* kb3 = (const float*)d_in[10];
    const float* kg3 = (const float*)d_in[11]; const float* kbe3= (const float*)d_in[12];
    const float* kw4 = (const float*)d_in[13]; const float* kb4 = (const float*)d_in[14];
    const float* pw1 = (const float*)d_in[15]; const float* pb1 = (const float*)d_in[16];
    const float* pg1 = (const float*)d_in[17]; const float* pbe1= (const float*)d_in[18];
    const float* pw2 = (const float*)d_in[19]; const float* pb2 = (const float*)d_in[20];
    const float* pg2 = (const float*)d_in[21]; const float* pbe2= (const float*)d_in[22];
    const float* pw3 = (const float*)d_in[23]; const float* pb3 = (const float*)d_in[24];

    float* out  = (float*)d_out;
    float* SRC  = out;
    float* KOUT = out + (size_t)1 * BB * HW;
    float* PSI  = out + (size_t)2 * BB * HW;
    float* AXO  = out + (size_t)3 * BB * HW;
    float* AYO  = out + (size_t)4 * BB * HW;

    float* WS    = (float*)d_ws;
    float* X16   = WS;                                   // B*16*HW
    float* BUF1  = X16  + (size_t)BB * 16 * HW;          // B*32*HW
    float* BUF2  = BUF1 + (size_t)BB * 32 * HW;          // B*32*HW
    float* STATS = BUF2 + (size_t)BB * 32 * HW;          // 2*B*8
    _Float16* WPACK = (_Float16*)(STATS + 2048);         // 16K halves max
    float* TMPB  = BUF1;                                 // reuse after branches
    float* PSIS  = BUF2;

    const int ELT = (BB * HW + 255) / 256;               // 4096 blocks
    const dim3 CGRID1(128, 1, BB), CGRID2(128, 2, BB);

    build_x16_kernel<<<ELT, 256, 0, stream>>>(image, X16);

    // ---- k branch ----
    pack_w_kernel<<<20, 256, 0, stream>>>(kw1, WPACK, 4, 16, 4, 2, 5);
    conv3x3_lds_wmma<16, 4, 32, 5, 11><<<CGRID2, 256, 0, stream>>>(X16, WPACK, kb1, BUF1);
    gn_stats_kernel<<<BB * 8, 256, 0, stream>>>(BUF1, STATS, 32, 8);
    gn_silu_kernel<<<(int)(((size_t)BB * 32 * HW + 255) / 256), 256, 0, stream>>>(
        BUF1, BUF2, kg1, kbe1, STATS, 32, 8);

    pack_w_kernel<<<36, 256, 0, stream>>>(kw2, WPACK, 32, 32, 5, 2, 9);
    conv3x3_lds_wmma<32, 5, 32, 9, 10><<<CGRID2, 256, 0, stream>>>(BUF2, WPACK, kb2, BUF1);
    gn_stats_kernel<<<BB * 8, 256, 0, stream>>>(BUF1, STATS, 32, 8);
    gn_silu_kernel<<<(int)(((size_t)BB * 32 * HW + 255) / 256), 256, 0, stream>>>(
        BUF1, BUF2, kg2, kbe2, STATS, 32, 8);

    pack_w_kernel<<<18, 256, 0, stream>>>(kw3, WPACK, 32, 32, 5, 1, 9);
    conv3x3_lds_wmma<32, 5, 16, 9, 10><<<CGRID1, 256, 0, stream>>>(BUF2, WPACK, kb3, BUF1);
    gn_stats_kernel<<<BB * 4, 256, 0, stream>>>(BUF1, STATS, 16, 4);
    gn_silu_kernel<<<(int)(((size_t)BB * 16 * HW + 255) / 256), 256, 0, stream>>>(
        BUF1, BUF2, kg3, kbe3, STATS, 16, 4);

    head_k_kernel<<<ELT, 256, 0, stream>>>(BUF2, kw4, kb4, KOUT);

    // ---- psi branch ----
    pack_w_kernel<<<10, 256, 0, stream>>>(pw1, WPACK, 4, 16, 4, 1, 5);
    conv3x3_lds_wmma<16, 4, 16, 5, 11><<<CGRID1, 256, 0, stream>>>(X16, WPACK, pb1, BUF1);
    gn_stats_kernel<<<BB * 4, 256, 0, stream>>>(BUF1, STATS, 16, 4);
    gn_silu_kernel<<<(int)(((size_t)BB * 16 * HW + 255) / 256), 256, 0, stream>>>(
        BUF1, BUF2, pg1, pbe1, STATS, 16, 4);

    pack_w_kernel<<<10, 256, 0, stream>>>(pw2, WPACK, 16, 16, 4, 1, 5);
    conv3x3_lds_wmma<16, 4, 16, 5, 11><<<CGRID1, 256, 0, stream>>>(BUF2, WPACK, pb2, BUF1);
    gn_stats_kernel<<<BB * 4, 256, 0, stream>>>(BUF1, STATS, 16, 4);
    gn_silu_kernel<<<(int)(((size_t)BB * 16 * HW + 255) / 256), 256, 0, stream>>>(
        BUF1, BUF2, pg2, pbe2, STATS, 16, 4);

    head_psi_kernel<<<ELT, 256, 0, stream>>>(BUF2, pw3, pb3, KOUT, PSI);

    // ---- blur + warp ----
    blur_h_kernel<<<ELT, 256, 0, stream>>>(PSI, TMPB);
    blur_v_kernel<<<ELT, 256, 0, stream>>>(TMPB, PSIS);
    final_kernel<<<ELT, 256, 0, stream>>>(PSIS, image, SRC, AXO, AYO);
}
